// HGAT_18133351924381
// MI455X (gfx1250) — matrix operand order repaired and checked by reading, not compiled, not run on hardware
//
#include <hip/hip_runtime.h>
#include <math.h>

// Problem constants (fixed by the reference)
#define N_NODES 1026
#define T_LEN   8
#define H_DIM   64
#define F_INPUT 5
#define E_EDGES 1026
#define NNZ_CNT 200000
#define NPAD    1040            // 65*16, row padding for WMMA tiles
#define NH      (N_NODES * H_DIM)
#define NHPAD   (NPAD * H_DIM)

typedef __attribute__((ext_vector_type(16))) _Float16     v16h;
typedef __attribute__((ext_vector_type(8)))  float        v8f;
typedef __attribute__((ext_vector_type(8)))  unsigned int v8u;

__device__ __forceinline__ float leaky_f(float x)   { return x >= 0.f ? x : 0.2f * x; }
__device__ __forceinline__ float sigmoid_f(float x) { return 1.f / (1.f + expf(-x)); }

// ---------------------------------------------------------------- utilities
__global__ void k_zero(float* __restrict__ p, int n) {
    int i = blockIdx.x * blockDim.x + threadIdx.x;
    if (i < n) p[i] = 0.f;
}

__global__ void k_invert(float* __restrict__ p, int n) {
    int i = blockIdx.x * blockDim.x + threadIdx.x;
    if (i < n) { float d = p[i]; p[i] = d > 0.f ? 1.f / d : 0.f; }
}

// fp32 [rows x 64] -> fp16 [rowspad x 64], zero-filled pad rows
__global__ void k_cvt_f16(const float* __restrict__ src, _Float16* __restrict__ dst,
                          int rows, int rowspad) {
    int i = blockIdx.x * blockDim.x + threadIdx.x;
    if (i >= rowspad * H_DIM) return;
    int r = i >> 6;
    dst[i] = (r < rows) ? (_Float16)src[i] : (_Float16)0.f;
}

// fp32 theta [64 x 64] -> fp16 transposed [64 x 64]: dst[n*64+k] = src[k*64+n]
__global__ void k_cvt_f16_tr(const float* __restrict__ src, _Float16* __restrict__ dst) {
    int i = blockIdx.x * blockDim.x + threadIdx.x;
    if (i >= H_DIM * H_DIM) return;
    int n = i >> 6, k = i & 63;
    dst[i] = (_Float16)src[k * H_DIM + n];
}

// ---------------------------------------------------------------- GRU
__global__ void k_gru(const float* __restrict__ x, const float* __restrict__ Wih,
                      const float* __restrict__ Whh, const float* __restrict__ bih,
                      const float* __restrict__ bhh, float* __restrict__ ctx,
                      float* __restrict__ last) {
    int n = blockIdx.x, j = threadIdx.x;
    __shared__ float h[H_DIM];
    h[j] = 0.f;
    __syncthreads();
    for (int t = 0; t < T_LEN; ++t) {
        const float* xt = x + (n * T_LEN + t) * F_INPUT;
        float gi[3], gh[3];
        #pragma unroll
        for (int p = 0; p < 3; ++p) {
            int k = p * H_DIM + j;
            float si = bih[k];
            #pragma unroll
            for (int f = 0; f < F_INPUT; ++f) si += xt[f] * Wih[k * F_INPUT + f];
            gi[p] = si;
            float sh = bhh[k];
            for (int d = 0; d < H_DIM; ++d) sh += h[d] * Whh[k * H_DIM + d];
            gh[p] = sh;
        }
        float r  = sigmoid_f(gi[0] + gh[0]);
        float z  = sigmoid_f(gi[1] + gh[1]);
        float nn = tanhf(gi[2] + r * gh[2]);
        float hn = (1.f - z) * nn + z * h[j];
        __syncthreads();
        h[j] = hn;
        __syncthreads();
        ctx[(n * T_LEN + t) * H_DIM + j] = hn;
    }
    last[n * H_DIM + j] = h[j];
}

// ---------------------------------------------------------------- decayed attention
__global__ void k_attn(const float* __restrict__ last, const float* __restrict__ ctx,
                       const float* __restrict__ Win, const float* __restrict__ Wout,
                       const float* __restrict__ ae, const float* __restrict__ ab,
                       float* __restrict__ out) {
    int n = blockIdx.x, j = threadIdx.x;
    __shared__ float ls[H_DIM], q[H_DIM], w[T_LEN], mixs[H_DIM];
    ls[j] = last[n * H_DIM + j];
    __syncthreads();
    float qj = 0.f;
    for (int d = 0; d < H_DIM; ++d) qj += ls[d] * Win[d * H_DIM + j];
    q[j] = qj;
    __syncthreads();
    if (j < T_LEN) {
        const float* c = ctx + (n * T_LEN + j) * H_DIM;
        float s = 0.f;
        for (int d = 0; d < H_DIM; ++d) s += q[d] * c[d];
        w[j] = s;
    }
    __syncthreads();
    if (j == 0) {
        float m = w[0];
        for (int t = 1; t < T_LEN; ++t) m = fmaxf(m, w[t]);
        float tot = 0.f;
        for (int t = 0; t < T_LEN; ++t) { w[t] = expf(w[t] - m); tot += w[t]; }
        for (int t = 0; t < T_LEN; ++t) w[t] /= tot;
    }
    __syncthreads();
    float ae_n = ae[n], ab_n = ab[n];
    float ms = 0.f;
    for (int t = 0; t < T_LEN; ++t) {
        float m  = w[t] * ctx[(n * T_LEN + t) * H_DIM + j];
        float bt = expf(-ab_n * (float)(T_LEN - 1 - t));
        float t2 = ae_n * m * bt;
        ms += (t2 > 0.f ? t2 : 0.f) + m;
    }
    mixs[j] = ms;
    __syncthreads();
    float o = 0.f;
    for (int k = 0; k < H_DIM; ++k) o += mixs[k] * Wout[k * H_DIM + j];
    for (int k = 0; k < H_DIM; ++k) o += q[k]    * Wout[(H_DIM + k) * H_DIM + j];
    out[n * H_DIM + j] = tanhf(o);
}

// ---------------------------------------------------------------- degrees
__global__ void k_count_deg(const int* __restrict__ hypT, const int* __restrict__ hyp,
                            float* __restrict__ degn, float* __restrict__ dege) {
    int tid = blockIdx.x * blockDim.x + threadIdx.x;
    if (tid >= 9 * NNZ_CNT) return;
    int g = tid / NNZ_CNT, i = tid - g * NNZ_CNT;
    int node, edge;
    if (g < 8) {
        node = hypT[g * 2 * NNZ_CNT + i];
        edge = hypT[g * 2 * NNZ_CNT + NNZ_CNT + i];
    } else {
        node = hyp[i];
        edge = hyp[NNZ_CNT + i];
    }
    atomicAdd(&degn[g * N_NODES + node], 1.f);
    atomicAdd(&dege[g * E_EDGES + edge], 1.f);
}

// ---------------------------------------------------------------- WMMA GEMM
// C[NPAD x 64] = A[NPAD x 64] * B,  B supplied transposed: BT[n][k] (64x64 f16).
// One wave computes a full 16x64 strip: A fragment loaded once, 4 N-tiles,
// 2 K-steps each -> 8 WMMAs. 4 waves per block.
__global__ void k_wmma_gemm(const _Float16* __restrict__ A, const _Float16* __restrict__ BT,
                            float* __restrict__ C) {
    int wave  = threadIdx.x >> 5;
    int lane  = threadIdx.x & 31;
    int mtile = blockIdx.x * 4 + wave;
    if (mtile >= NPAD / 16) return;
    int m0  = mtile * 16;
    int r15 = lane & 15, hi = lane >> 4;
    int row = m0 + r15;

    // A fragments for k0=0 and k0=32 (16-bit A 16x32 layout; pairs are contiguous)
    v8u au0, au1;
    const _Float16* Ar = A + row * 64;
    #pragma unroll
    for (int v = 0; v < 8; ++v) {
        int kk = ((v < 4) ? (2 * v) : (16 + 2 * (v - 4))) + 8 * hi;
        au0[v] = *(const unsigned int*)(Ar + kk);
        au1[v] = *(const unsigned int*)(Ar + 32 + kk);
    }
    v16h a0 = __builtin_bit_cast(v16h, au0);
    v16h a1 = __builtin_bit_cast(v16h, au1);

    #pragma unroll
    for (int nt = 0; nt < 4; ++nt) {
        int col = nt * 16 + r15;
        const _Float16* Br = BT + col * 64;   // BT row = column of B, contiguous in k
        v8u bu0, bu1;
        #pragma unroll
        for (int v = 0; v < 8; ++v) {
            int k = ((v < 4) ? (2 * v) : (16 + 2 * (v - 4))) + 8 * hi;   // B 32x16 layout: same k order as A
            bu0[v] = *(const unsigned int*)(Br + k);
            bu1[v] = *(const unsigned int*)(Br + 32 + k);
        }
        v8f c = {};
        c = __builtin_amdgcn_wmma_f32_16x16x32_f16(false, a0, false,
                __builtin_bit_cast(v16h, bu0), (short)0, c, false, false);
        c = __builtin_amdgcn_wmma_f32_16x16x32_f16(false, a1, false,
                __builtin_bit_cast(v16h, bu1), (short)0, c, false, false);
        #pragma unroll
        for (int r = 0; r < 8; ++r)
            C[(m0 + r + 8 * hi) * 64 + col] = c[r];   // unconditional: C padded
    }
}

// ---------------------------------------------------------------- hypergraph scatter/gather
// 4 H-elements per thread: one b128 load + 4 atomics, index/Binv amortized 4x.
__global__ void k_scatter_e(const float* __restrict__ xt, const int* __restrict__ nodep,
                            const int* __restrict__ edgep, const float* __restrict__ binv,
                            float* __restrict__ ebuf) {
    int tid = blockIdx.x * blockDim.x + threadIdx.x;
    if (tid >= NNZ_CNT * 16) return;
    int i = tid >> 4, j = (tid & 15) << 2;
    int node = nodep[i], edge = edgep[i];
    float bi = binv[edge];
    const float4 v = *(const float4*)(xt + node * H_DIM + j);
    float* e = ebuf + edge * H_DIM + j;
    atomicAdd(e + 0, v.x * bi);
    atomicAdd(e + 1, v.y * bi);
    atomicAdd(e + 2, v.z * bi);
    atomicAdd(e + 3, v.w * bi);
}

__global__ void k_gather_n(const float* __restrict__ ebuf, const int* __restrict__ nodep,
                           const int* __restrict__ edgep, float* __restrict__ acc) {
    int tid = blockIdx.x * blockDim.x + threadIdx.x;
    if (tid >= NNZ_CNT * 16) return;
    int i = tid >> 4, j = (tid & 15) << 2;
    const float4 v = *(const float4*)(ebuf + edgep[i] * H_DIM + j);
    float* a = acc + nodep[i] * H_DIM + j;
    atomicAdd(a + 0, v.x);
    atomicAdd(a + 1, v.y);
    atomicAdd(a + 2, v.z);
    atomicAdd(a + 3, v.w);
}

__global__ void k_finalize(const float* __restrict__ acc, const float* __restrict__ dinv,
                           const float* __restrict__ bias, float* __restrict__ out) {
    int tid = blockIdx.x * blockDim.x + threadIdx.x;
    if (tid >= NH) return;
    int n = tid >> 6, j = tid & 63;
    out[tid] = leaky_f(acc[tid] * dinv[n] + bias[j]);
}

// ---------------------------------------------------------------- temporal reductions
__global__ void k_sub_sum(const float* __restrict__ x2, float* __restrict__ bsum) {
    int t = blockIdx.y;
    int i = blockIdx.x * blockDim.x + threadIdx.x;
    float s = 0.f;
    if (i < NH) s = x2[(t + 1) * NH + i] - x2[t * NH + i];
    __shared__ float red[256];
    red[threadIdx.x] = s;
    __syncthreads();
    for (int st = 128; st > 0; st >>= 1) {
        if (threadIdx.x < st) red[threadIdx.x] += red[threadIdx.x + st];
        __syncthreads();
    }
    if (threadIdx.x == 0) atomicAdd(&bsum[t], red[0]);
}

__global__ void k_scale(float* __restrict__ p, int n, float s) {
    int i = blockIdx.x * blockDim.x + threadIdx.x;
    if (i < n) p[i] *= s;
}

__global__ void k_z_sum(const float* __restrict__ x2, const float* __restrict__ bmean,
                        const float* __restrict__ a_param, float* __restrict__ zsum) {
    int t = blockIdx.y;
    int i = blockIdx.x * blockDim.x + threadIdx.x;
    float val = 0.f;
    if (i < NH) {
        float s = x2[(t + 1) * NH + i] - x2[t * NH + i];
        float a = a_param[t];
        float U = 1.f / (1.f + a * (bmean[t] - s));
        val = (U * s) / U;       // faithful to reference
    }
    __shared__ float red[256];
    red[threadIdx.x] = val;
    __syncthreads();
    for (int st = 128; st > 0; st >>= 1) {
        if (threadIdx.x < st) red[threadIdx.x] += red[threadIdx.x + st];
        __syncthreads();
    }
    if (threadIdx.x == 0) atomicAdd(&zsum[t], red[0]);
}

__global__ void k_wattn(const float* __restrict__ w1, const float* __restrict__ w2,
                        const float* __restrict__ z, float* __restrict__ wat) {
    __shared__ float hv[H_DIM], zz[7], s2[7];
    int j = threadIdx.x;
    if (j < 7) zz[j] = z[j];
    __syncthreads();
    float s = 0.f;
    for (int t = 0; t < 7; ++t) s += w1[j * 7 + t] * zz[t];
    hv[j] = leaky_f(s);
    __syncthreads();
    if (j < 7) {
        float r = 0.f;
        for (int k = 0; k < H_DIM; ++k) r += w2[j * H_DIM + k] * hv[k];
        s2[j] = r;
    }
    __syncthreads();
    if (j == 0) {
        float m = s2[0];
        for (int t = 1; t < 7; ++t) m = fmaxf(m, s2[t]);
        float tot = 0.f, e[7];
        for (int t = 0; t < 7; ++t) { e[t] = expf(s2[t] - m); tot += e[t]; }
        for (int t = 0; t < 7; ++t) wat[t] = e[t] / tot;
    }
}

__global__ void k_final(const float* __restrict__ xg, const float* __restrict__ x2,
                        const float* __restrict__ wat, const float* __restrict__ Wl,
                        const float* __restrict__ bl, float* __restrict__ out) {
    int n = blockIdx.x * blockDim.x + threadIdx.x;
    if (n >= N_NODES) return;
    float w0 = wat[0], w2v = wat[2];
    float acc = bl[0];
    const float* g  = xg + n * H_DIM;
    const float* a0 = x2 + 0 * NH + n * H_DIM;  // sub0 = x2[1]-x2[0]
    const float* a1 = x2 + 1 * NH + n * H_DIM;
    const float* a2 = x2 + 2 * NH + n * H_DIM;  // sub2 = x2[3]-x2[2]
    const float* a3 = x2 + 3 * NH + n * H_DIM;
    for (int j = 0; j < H_DIM; ++j) acc += g[j] * Wl[j];
    for (int j = 0; j < H_DIM; ++j) {
        float xx = w0 * (a1[j] - a0[j]) + w2v * (a3[j] - a2[j]);
        acc += xx * Wl[H_DIM + j];
    }
    out[n] = leaky_f(acc);
}

// ---------------------------------------------------------------- host launch
extern "C" void kernel_launch(void* const* d_in, const int* in_sizes, int n_in,
                              void* d_out, int out_size, void* d_ws, size_t ws_size,
                              hipStream_t stream) {
    const float* price = (const float*)d_in[0];
    const int*   hypT  = (const int*)  d_in[1];
    const int*   hyp   = (const int*)  d_in[2];
    const float* Wih   = (const float*)d_in[3];
    const float* Whh   = (const float*)d_in[4];
    const float* bih   = (const float*)d_in[5];
    const float* bhh   = (const float*)d_in[6];
    const float* Win   = (const float*)d_in[7];
    const float* Wout  = (const float*)d_in[8];
    const float* ae    = (const float*)d_in[9];
    const float* ab    = (const float*)d_in[10];
    const float* th1   = (const float*)d_in[11];
    const float* b1    = (const float*)d_in[12];
    const float* th2   = (const float*)d_in[13];
    const float* b2    = (const float*)d_in[14];
    const float* w1    = (const float*)d_in[15];
    const float* w2    = (const float*)d_in[16];
    const float* a_p   = (const float*)d_in[17];
    const float* Wl    = (const float*)d_in[18];
    const float* bl    = (const float*)d_in[19];
    float* out = (float*)d_out;

    // scratch carve-out
    char* base = (char*)d_ws;
    size_t off = 0;
    auto carve = [&](size_t bytes) -> void* {
        off = (off + 255) & ~(size_t)255;
        void* p = base + off;
        off += bytes;
        return p;
    };
    float*    ctx   = (float*)   carve(sizeof(float) * N_NODES * T_LEN * H_DIM);
    float*    last  = (float*)   carve(sizeof(float) * NH);
    float*    attn  = (float*)   carve(sizeof(float) * NH);
    float*    degn  = (float*)   carve(sizeof(float) * 9 * N_NODES);
    float*    dege  = (float*)   carve(sizeof(float) * 9 * E_EDGES);
    float*    XT1   = (float*)   carve(sizeof(float) * NHPAD);      // padded GEMM out
    float*    XT2   = (float*)   carve(sizeof(float) * NHPAD);      // padded GEMM out
    float*    x1    = (float*)   carve(sizeof(float) * NH);
    float*    ebuf  = (float*)   carve(sizeof(float) * E_EDGES * H_DIM);
    float*    acc   = (float*)   carve(sizeof(float) * NH);
    float*    x2    = (float*)   carve(sizeof(float) * T_LEN * NH);
    float*    xg    = (float*)   carve(sizeof(float) * NH);
    _Float16* Ah    = (_Float16*)carve(sizeof(_Float16) * NHPAD);
    _Float16* th1hT = (_Float16*)carve(sizeof(_Float16) * H_DIM * H_DIM);
    _Float16* th2hT = (_Float16*)carve(sizeof(_Float16) * H_DIM * H_DIM);
    float*    bsum  = (float*)   carve(sizeof(float) * 7);
    float*    zsum  = (float*)   carve(sizeof(float) * 7);
    float*    wat   = (float*)   carve(sizeof(float) * 7);
    (void)ws_size; (void)n_in; (void)in_sizes; (void)out_size;

    auto cdiv = [](int a, int b) { return (a + b - 1) / b; };
    const int nhBlk  = cdiv(NH, 256);
    const int nnzBlk = cdiv(NNZ_CNT * 16, 256);
    const int gemmGrid = cdiv(NPAD / 16, 4);   // 4 waves (m-tiles) per block

    // 1) GRU + attention
    k_gru <<<N_NODES, H_DIM, 0, stream>>>(price, Wih, Whh, bih, bhh, ctx, last);
    k_attn<<<N_NODES, H_DIM, 0, stream>>>(last, ctx, Win, Wout, ae, ab, attn);

    // 2) degrees for all 9 hypergraphs
    k_zero<<<cdiv(9 * N_NODES, 256), 256, 0, stream>>>(degn, 9 * N_NODES);
    k_zero<<<cdiv(9 * E_EDGES, 256), 256, 0, stream>>>(dege, 9 * E_EDGES);
    k_count_deg<<<cdiv(9 * NNZ_CNT, 256), 256, 0, stream>>>(hypT, hyp, degn, dege);
    k_invert<<<cdiv(9 * N_NODES, 256), 256, 0, stream>>>(degn, 9 * N_NODES);
    k_invert<<<cdiv(9 * E_EDGES, 256), 256, 0, stream>>>(dege, 9 * E_EDGES);

    // 3) fp16 weights (transposed) + shared layer-1 transform XT1 = attn @ theta1
    k_cvt_f16_tr<<<cdiv(H_DIM * H_DIM, 256), 256, 0, stream>>>(th1, th1hT);
    k_cvt_f16_tr<<<cdiv(H_DIM * H_DIM, 256), 256, 0, stream>>>(th2, th2hT);
    k_cvt_f16<<<cdiv(NHPAD, 256), 256, 0, stream>>>(attn, Ah, N_NODES, NPAD);
    k_wmma_gemm<<<gemmGrid, 128, 0, stream>>>(Ah, th1hT, XT1);

    // 4) 9 graphs: 8 temporal (-> x2[t]) + 1 global (-> xg)
    for (int g = 0; g < 9; ++g) {
        const int* nodep = (g < 8) ? (hypT + g * 2 * NNZ_CNT) : hyp;
        const int* edgep = (g < 8) ? (hypT + g * 2 * NNZ_CNT + NNZ_CNT) : (hyp + NNZ_CNT);
        const float* dinv = degn + g * N_NODES;
        const float* binv = dege + g * E_EDGES;
        float* dst = (g < 8) ? (x2 + g * NH) : xg;

        // layer 1: D^-1 H B^-1 H^T XT1 + b1, leaky
        k_zero<<<nhBlk, 256, 0, stream>>>(ebuf, E_EDGES * H_DIM);
        k_scatter_e<<<nnzBlk, 256, 0, stream>>>(XT1, nodep, edgep, binv, ebuf);
        k_zero<<<nhBlk, 256, 0, stream>>>(acc, NH);
        k_gather_n<<<nnzBlk, 256, 0, stream>>>(ebuf, nodep, edgep, acc);
        k_finalize<<<nhBlk, 256, 0, stream>>>(acc, dinv, b1, x1);

        // layer 2: XT2 = x1 @ theta2 (WMMA), then conv, leaky
        k_cvt_f16<<<cdiv(NHPAD, 256), 256, 0, stream>>>(x1, Ah, N_NODES, NPAD);
        k_wmma_gemm<<<gemmGrid, 128, 0, stream>>>(Ah, th2hT, XT2);
        k_zero<<<nhBlk, 256, 0, stream>>>(ebuf, E_EDGES * H_DIM);
        k_scatter_e<<<nnzBlk, 256, 0, stream>>>(XT2, nodep, edgep, binv, ebuf);
        k_zero<<<nhBlk, 256, 0, stream>>>(acc, NH);
        k_gather_n<<<nnzBlk, 256, 0, stream>>>(ebuf, nodep, edgep, acc);
        k_finalize<<<nhBlk, 256, 0, stream>>>(acc, dinv, b2, dst);
    }

    // 5) temporal attention weights
    k_zero<<<1, 32, 0, stream>>>(bsum, 7);
    k_sub_sum<<<dim3(nhBlk, 7), 256, 0, stream>>>(x2, bsum);
    k_scale<<<1, 32, 0, stream>>>(bsum, 7, 1.f / (float)NH);     // bsum -> b (mean)
    k_zero<<<1, 32, 0, stream>>>(zsum, 7);
    k_z_sum<<<dim3(nhBlk, 7), 256, 0, stream>>>(x2, bsum, a_p, zsum);
    k_wattn<<<1, H_DIM, 0, stream>>>(w1, w2, zsum, wat);

    // 6) final projection
    k_final<<<cdiv(N_NODES, 128), 128, 0, stream>>>(xg, x2, wat, Wl, bl, out);
}
